// TransformerEncoder_66778151518244
// MI455X (gfx1250) — matrix operand order repaired and checked by reading, not compiled, not run on hardware
//
#include <hip/hip_runtime.h>

// ---------------- problem constants (from reference) ----------------
constexpr int Lc  = 6;
constexpr int Dc  = 1024;
constexpr int Hc  = 16;
constexpr int Fc  = 4096;
constexpr int Bc  = 2;
constexpr int Sc  = 1024;
constexpr int DKc = Dc / Hc;      // 64
constexpr int Mc  = Bc * Sc;      // 2048 tokens
constexpr float EPSc = 1e-6f;

// ---------------- vector types ----------------
typedef __attribute__((ext_vector_type(16))) __bf16 v16bf;
typedef __attribute__((ext_vector_type(8)))  float  v8f;
typedef __attribute__((ext_vector_type(4)))  float  v4f;
typedef __attribute__((ext_vector_type(2)))  float  v2f;
typedef __attribute__((ext_vector_type(8)))  unsigned v8u;

__device__ __forceinline__ v8f wmma_bf16(v16bf a, v16bf b, v8f c) {
  // D = A(16x32 bf16) * B(32x16 bf16) + C(16x16 f32)
  return __builtin_amdgcn_wmma_f32_16x16x32_bf16(
      /*neg_a=*/false, a, /*neg_b=*/false, b,
      /*c_mod=*/(short)0, c, /*reuse_a=*/false, /*reuse_b=*/false);
}

__device__ __forceinline__ __bf16 f2bf(float f) { return (__bf16)f; }

// packed f32x2 -> bf16x2 in one instruction (backend won't emit this itself)
__device__ __forceinline__ unsigned cvtpk(float x, float y) {
  unsigned d;
  asm("v_cvt_pk_bf16_f32 %0, %1, %2" : "=v"(d) : "v"(x), "v"(y));
  return d;   // lo = bf16(x), hi = bf16(y)
}

__device__ __forceinline__ v16bf u8_as_bf16(v8u u) {
  v16bf r;
  __builtin_memcpy(&r, &u, 32);
  return r;
}

// 16 contiguous f32 -> v16bf via 8 packed converts
__device__ __forceinline__ v16bf cvt16(const float* __restrict__ p) {
  v8f a = *(const v8f*)p;
  v8f b = *(const v8f*)(p + 8);
  v8u u;
  #pragma unroll
  for (int i = 0; i < 4; ++i) u[i]     = cvtpk(a[2*i], a[2*i+1]);
  #pragma unroll
  for (int i = 0; i < 4; ++i) u[4 + i] = cvtpk(b[2*i], b[2*i+1]);
  return u8_as_bf16(u);
}

// two groups of 8 contiguous f32 -> v16bf (A-fragment pattern)
__device__ __forceinline__ v16bf cvt8x2(const float* __restrict__ p0,
                                        const float* __restrict__ p1) {
  v8f a = *(const v8f*)p0;
  v8f b = *(const v8f*)p1;
  v8u u;
  #pragma unroll
  for (int i = 0; i < 4; ++i) u[i]     = cvtpk(a[2*i], a[2*i+1]);
  #pragma unroll
  for (int i = 0; i < 4; ++i) u[4 + i] = cvtpk(b[2*i], b[2*i+1]);
  return u8_as_bf16(u);
}

// swap bits 3<->4 of a K index: stores A panels in fragment order so each
// lane's 16 A elements are one contiguous 32B LDS read.
__device__ __forceinline__ int kperm(int k) {
  return (k & ~24) | ((k & 8) << 1) | ((k & 16) >> 1);
}

// =====================================================================
// Embedding: x[tok,:] = emb[ids[tok],:] * sqrt(D) + pe[tok % S, :]
// =====================================================================
__global__ __launch_bounds__(256) void te_embed(const int* __restrict__ ids,
                                                const float* __restrict__ emb,
                                                const float* __restrict__ pe,
                                                float* __restrict__ x) {
  const int tok = blockIdx.x;
  const int sp  = tok & (Sc - 1);
  const int id  = ids[tok];
  const float scale = 32.0f;           // sqrt(1024)
  for (int i = threadIdx.x; i < Dc; i += 256) {
    x[(size_t)tok * Dc + i] = emb[(size_t)id * Dc + i] * scale + pe[(size_t)sp * Dc + i];
  }
}

// =====================================================================
// LayerNorm: one block per token row (D = 1024)
// =====================================================================
__global__ __launch_bounds__(256) void te_ln(const float* __restrict__ x,
                                             const float* __restrict__ sc,
                                             const float* __restrict__ bi,
                                             float* __restrict__ out) {
  const int row = blockIdx.x;
  const float* xr = x + (size_t)row * Dc;
  float lsum = 0.f, lsq = 0.f;
  for (int i = threadIdx.x; i < Dc; i += 256) {
    float v = xr[i];
    lsum += v;
    lsq  += v * v;
  }
  __shared__ float red[2][8];
  #pragma unroll
  for (int off = 16; off > 0; off >>= 1) {
    lsum += __shfl_xor(lsum, off, 32);
    lsq  += __shfl_xor(lsq,  off, 32);
  }
  const int wv = threadIdx.x >> 5;
  if ((threadIdx.x & 31) == 0) { red[0][wv] = lsum; red[1][wv] = lsq; }
  __syncthreads();
  if (threadIdx.x < 32) {
    float s0 = (threadIdx.x < 8) ? red[0][threadIdx.x] : 0.f;
    float s1 = (threadIdx.x < 8) ? red[1][threadIdx.x] : 0.f;
    #pragma unroll
    for (int off = 4; off > 0; off >>= 1) {
      s0 += __shfl_xor(s0, off, 32);
      s1 += __shfl_xor(s1, off, 32);
    }
    if (threadIdx.x == 0) { red[0][0] = s0; red[1][0] = s1; }
  }
  __syncthreads();
  const float mean = red[0][0] * (1.0f / Dc);
  const float var  = red[1][0] * (1.0f / Dc) - mean * mean;
  const float rinv = rsqrtf(var + EPSc);
  for (int i = threadIdx.x; i < Dc; i += 256) {
    out[(size_t)row * Dc + i] = (xr[i] - mean) * rinv * sc[i] + bi[i];
  }
}

// =====================================================================
// GEMM: C = act( A(MxK) @ W(KxN) + bias ) + res
//   256 threads = 8 waves in a 2x4 grid of 32x32 subtiles -> 64x128 tile.
//   Double-buffered LDS staging: stage panel k+32 while computing panel k,
//   one barrier per k-step. A panel in fragment order [m][kperm(k)],
//   B panel transposed [n][kperm(k)]; every fragment is one contiguous 32B read.
//   4 WMMAs per wave per 32-deep k-step.
// =====================================================================
template <bool RELU, bool BIAS, bool RES>
__global__ __launch_bounds__(256) void te_gemm(const float* __restrict__ A,
                                               const float* __restrict__ W,
                                               const float* __restrict__ bias,
                                               const float* __restrict__ res,
                                               float* __restrict__ C,
                                               int M, int N, int K) {
  __shared__ __align__(16) __bf16 As[2][64][40];    // ping-pong, pitch 80B
  __shared__ __align__(16) __bf16 Bt[2][128][40];   // ping-pong, pitch 80B

  const int tid  = threadIdx.x;
  const int lane = tid & 31;
  const int wave = tid >> 5;          // 0..7
  const int wr   = wave >> 2;         // 0..1 : 32-row group
  const int wc   = wave & 3;          // 0..3 : 32-col group
  const int row0 = blockIdx.x * 64;
  const int col0 = blockIdx.y * 128;
  const int r    = lane & 15;
  const int lh   = lane >> 4;         // lane-half

  auto stage = [&](int buf, int kbase) {
    // A panel (64 x 32): f32x2 load -> one packed cvt -> b32 LDS store
    #pragma unroll
    for (int t = 0; t < 4; ++t) {                 // 64*16 pairs / 256 threads
      const int i  = tid + t * 256;
      const int m  = i >> 4;
      const int kk = (i & 15) * 2;
      const float* ap = &A[(size_t)(row0 + m) * K + (kbase + kk)];
      v2f f = *(const v2f*)ap;
      *(unsigned*)&As[buf][m][kperm(kk)] = cvtpk(f.x, f.y);
      if (kbase + 64 <= K - 32) __builtin_prefetch(ap + 32, 0, 1);  // k+64 panel
    }
    // B panel (32 x 128) transposed as [n][kperm(k)]
    #pragma unroll
    for (int t = 0; t < 8; ++t) {                 // 16*128 pairs / 256 threads
      const int i  = tid + t * 256;
      const int kk = (i >> 7) * 2;
      const int n  = i & 127;
      const float* wp = &W[(size_t)(kbase + kk) * N + (col0 + n)];
      float f0 = wp[0];
      float f1 = wp[N];
      *(unsigned*)&Bt[buf][n][kperm(kk)] = cvtpk(f0, f1);
      if (kbase + 64 <= K - 32) __builtin_prefetch(wp + (size_t)32 * N, 0, 1);
    }
  };

  v8f acc[2][2] = {};

  stage(0, 0);
  __syncthreads();

  int cur = 0;
  for (int k0 = 0; k0 < K; k0 += 32) {
    const int nxt = cur ^ 1;
    if (k0 + 32 < K) stage(nxt, k0 + 32);   // overlap with compute below

    // ---- fragments from current buffer: contiguous 32B LDS reads ----
    v16bf af[2], bfr[2];
    #pragma unroll
    for (int i = 0; i < 2; ++i) {
      const int am = wr * 32 + i * 16 + r;
      __builtin_memcpy(&af[i], &As[cur][am][lh * 16], 32);
      const int bn = wc * 32 + i * 16 + r;
      __builtin_memcpy(&bfr[i], &Bt[cur][bn][lh * 16], 32);
    }
    acc[0][0] = wmma_bf16(af[0], bfr[0], acc[0][0]);
    acc[0][1] = wmma_bf16(af[0], bfr[1], acc[0][1]);
    acc[1][0] = wmma_bf16(af[1], bfr[0], acc[1][0]);
    acc[1][1] = wmma_bf16(af[1], bfr[1], acc[1][1]);

    __syncthreads();    // next buffer fully staged; current reads retired
    cur = nxt;
  }

  // ---- epilogue. C elem e -> row = base + lh*8 + e, col = base + r ----
  #pragma unroll
  for (int j = 0; j < 2; ++j) {
    const int ccol = col0 + wc * 32 + j * 16 + r;
    const float bv = BIAS ? bias[ccol] : 0.f;
    #pragma unroll
    for (int i = 0; i < 2; ++i) {
      #pragma unroll
      for (int e = 0; e < 8; ++e) {
        const int crow = row0 + wr * 32 + i * 16 + lh * 8 + e;
        float v = acc[i][j][e] + bv;
        if (RELU) v = fmaxf(v, 0.f);
        if (RES) v += res[(size_t)crow * N + ccol];
        C[(size_t)crow * N + ccol] = v;
      }
    }
  }
}

// =====================================================================
// Flash attention: one wave per (b, h, 16-row q tile).
//   32-key chunks; scores via 2x WMMA bf16, online softmax in registers,
//   P staged in fragment order, V chunk staged transposed; all WMMA
//   fragments are single contiguous 32B LDS reads or packed-cvt builds.
// =====================================================================
__global__ __launch_bounds__(32) void te_attn(const float* __restrict__ Qb,
                                              const float* __restrict__ Kb,
                                              const float* __restrict__ Vb,
                                              const float* __restrict__ mask,
                                              float* __restrict__ Ob) {
  const int lane = threadIdx.x;
  const int qt   = blockIdx.x;           // q tile (0..S/16-1)
  const int h    = blockIdx.y;           // head
  const int b    = blockIdx.z;           // batch
  const int hoff = h * DKc;
  const size_t base = (size_t)b * Sc * Dc;
  const int r  = lane & 15;
  const int lh = lane >> 4;
  const float scale = 0.125f;            // 1/sqrt(64)

  __shared__ __align__(16) __bf16 Pl[16][40];   // P tile in fragment order
  __shared__ __align__(16) __bf16 Vt[64][40];   // V chunk transposed [dim][key]

  // P store positions (key-in-chunk r and 16+r, contiguous), fixed per lane
  const int pos0 = r;
  const int pos1 = 16 + r;

  // Q fragments (A layout) for kdim chunks [0,32) and [32,64)
  v16bf qa[2];
  const int qrow = qt * 16 + r;
  #pragma unroll
  for (int c = 0; c < 2; ++c) {
    const float* qp = Qb + base + (size_t)qrow * Dc + hoff + c * 32;
    qa[c] = cvt16(qp + lh * 16);
  }

  float mrow[8], lrow[8];
  #pragma unroll
  for (int e = 0; e < 8; ++e) { mrow[e] = -3.0e38f; lrow[e] = 0.f; }
  v8f o[4] = {};                          // 16x64 output accumulator

  for (int kc = 0; kc < Sc; kc += 32) {
    // prefetch next chunk's K/V rows (one row per lane)
    if (kc + 32 < Sc) {
      __builtin_prefetch(Kb + base + (size_t)(kc + 32 + lane) * Dc + hoff, 0, 1);
      __builtin_prefetch(Vb + base + (size_t)(kc + 32 + lane) * Dc + hoff, 0, 1);
    }

    // ---- stage V chunk transposed: lane owns key row (kc+lane) ----
    {
      const float* vp = Vb + base + (size_t)(kc + lane) * Dc + hoff;
      #pragma unroll
      for (int d4 = 0; d4 < 16; ++d4) {
        v4f f = ((const v4f*)vp)[d4];
        Vt[d4 * 4 + 0][lane] = f2bf(f.x);
        Vt[d4 * 4 + 1][lane] = f2bf(f.y);
        Vt[d4 * 4 + 2][lane] = f2bf(f.z);
        Vt[d4 * 4 + 3][lane] = f2bf(f.w);
      }
    }

    // ---- scores: two 16-key groups ----
    v8f scv[2];
    #pragma unroll
    for (int g = 0; g < 2; ++g) {
      const int key = kc + g * 16 + r;    // B column = key index
      const float* kp = Kb + base + (size_t)key * Dc + hoff;
      v16bf bk0 = cvt16(kp + lh * 16);
      v16bf bk1 = cvt16(kp + 32 + lh * 16);
      v8f z = {};
      z      = wmma_bf16(qa[0], bk0, z);
      scv[g] = wmma_bf16(qa[1], bk1, z);
    }
    // scale + mask (key = kc + g*16 + (lane&15))
    const float mk0 = mask[b * Sc + kc + r];
    const float mk1 = mask[b * Sc + kc + 16 + r];
    #pragma unroll
    for (int e = 0; e < 8; ++e) {
      scv[0][e] = (mk0 == 0.f) ? -1.0e9f : scv[0][e] * scale;
      scv[1][e] = (mk1 == 0.f) ? -1.0e9f : scv[1][e] * scale;
    }
    // ---- online softmax ----
    #pragma unroll
    for (int e = 0; e < 8; ++e) {
      float v = fmaxf(scv[0][e], scv[1][e]);
      #pragma unroll
      for (int off = 1; off < 16; off <<= 1) v = fmaxf(v, __shfl_xor(v, off, 32));
      const float newm  = fmaxf(mrow[e], v);
      const float alpha = __expf(mrow[e] - newm);
      lrow[e] *= alpha;
      #pragma unroll
      for (int t = 0; t < 4; ++t) o[t][e] *= alpha;
      const float p0 = __expf(scv[0][e] - newm);
      const float p1 = __expf(scv[1][e] - newm);
      float rs = p0 + p1;
      #pragma unroll
      for (int off = 1; off < 16; off <<= 1) rs += __shfl_xor(rs, off, 32);
      lrow[e] += rs;
      mrow[e]  = newm;
      // stage P in fragment order (row = e + 8*lh)
      Pl[e + lh * 8][pos0] = f2bf(p0);
      Pl[e + lh * 8][pos1] = f2bf(p1);
    }
    asm volatile("s_wait_dscnt 0" ::: "memory");   // stores visible cross-lane

    // ---- P (16x32, A layout) @ V (32x64) ----
    v16bf pa;
    __builtin_memcpy(&pa, &Pl[r][lh * 16], 32);    // one contiguous 32B read
    #pragma unroll
    for (int t = 0; t < 4; ++t) {
      v16bf bv;
      __builtin_memcpy(&bv, &Vt[t * 16 + r][lh * 16], 32);
      o[t] = wmma_bf16(pa, bv, o[t]);
    }
    asm volatile("s_wait_dscnt 0" ::: "memory");   // reads done before next stores
  }

  // ---- finalize: O /= l ----
  #pragma unroll
  for (int e = 0; e < 8; ++e) {
    const float inv = 1.0f / lrow[e];
    const int row = qt * 16 + lh * 8 + e;
    #pragma unroll
    for (int t = 0; t < 4; ++t)
      Ob[base + (size_t)row * Dc + hoff + t * 16 + r] = o[t][e] * inv;
  }
}

// =====================================================================
// Host-side orchestration
// =====================================================================
extern "C" void kernel_launch(void* const* d_in, const int* in_sizes, int n_in,
                              void* d_out, int out_size, void* d_ws, size_t ws_size,
                              hipStream_t stream) {
  (void)in_sizes; (void)n_in; (void)out_size; (void)ws_size;

  const int*   ids  = (const int*)  d_in[0];
  const float* mask = (const float*)d_in[1];
  const float* emb  = (const float*)d_in[2];
  const float* pe   = (const float*)d_in[3];
  const float* w_q  = (const float*)d_in[4];
  const float* w_k  = (const float*)d_in[5];
  const float* w_v  = (const float*)d_in[6];
  const float* w_o  = (const float*)d_in[7];
  const float* b_o  = (const float*)d_in[8];
  const float* w1   = (const float*)d_in[9];
  const float* b1   = (const float*)d_in[10];
  const float* w2   = (const float*)d_in[11];
  const float* b2   = (const float*)d_in[12];
  const float* ln1s = (const float*)d_in[13];
  const float* ln1b = (const float*)d_in[14];
  const float* ln2s = (const float*)d_in[15];
  const float* ln2b = (const float*)d_in[16];
  const float* lnfs = (const float*)d_in[17];
  const float* lnfb = (const float*)d_in[18];

  const size_t MD = (size_t)Mc * Dc;      // 2M floats
  float* x  = (float*)d_ws;
  float* hb = x  + MD;
  float* qb = hb + MD;
  float* kb = qb + MD;
  float* vb = kb + MD;
  float* ao = vb + MD;
  float* f1 = ao + MD;                     // Mc x Fc

  te_embed<<<Mc, 256, 0, stream>>>(ids, emb, pe, x);

  const dim3 gD(Mc / 64, Dc / 128);        // 32 x 8
  const dim3 gF(Mc / 64, Fc / 128);        // 32 x 32
  const dim3 gAttn(Sc / 16, Hc, Bc);       // 64 x 16 x 2

  for (int l = 0; l < Lc; ++l) {
    te_ln<<<Mc, 256, 0, stream>>>(x, ln1s + (size_t)l * Dc, ln1b + (size_t)l * Dc, hb);

    te_gemm<false, false, false><<<gD, 256, 0, stream>>>(
        hb, w_q + (size_t)l * Dc * Dc, nullptr, nullptr, qb, Mc, Dc, Dc);
    te_gemm<false, false, false><<<gD, 256, 0, stream>>>(
        hb, w_k + (size_t)l * Dc * Dc, nullptr, nullptr, kb, Mc, Dc, Dc);
    te_gemm<false, false, false><<<gD, 256, 0, stream>>>(
        hb, w_v + (size_t)l * Dc * Dc, nullptr, nullptr, vb, Mc, Dc, Dc);

    te_attn<<<gAttn, 32, 0, stream>>>(qb, kb, vb, mask, ao);

    // x = x + ao @ w_o + b_o
    te_gemm<false, true, true><<<gD, 256, 0, stream>>>(
        ao, w_o + (size_t)l * Dc * Dc, b_o + (size_t)l * Dc, x, x, Mc, Dc, Dc);

    te_ln<<<Mc, 256, 0, stream>>>(x, ln2s + (size_t)l * Dc, ln2b + (size_t)l * Dc, hb);

    // f1 = relu(hb @ w1 + b1)
    te_gemm<true, true, false><<<gF, 256, 0, stream>>>(
        hb, w1 + (size_t)l * Dc * Fc, b1 + (size_t)l * Fc, nullptr, f1, Mc, Fc, Dc);
    // x = x + f1 @ w2 + b2
    te_gemm<false, true, true><<<gD, 256, 0, stream>>>(
        f1, w2 + (size_t)l * Fc * Dc, b2 + (size_t)l * Dc, x, x, Mc, Dc, Fc);
  }

  te_ln<<<Mc, 256, 0, stream>>>(x, lnfs, lnfb, (float*)d_out);
}
